// MultiHeadAttention_52785148068514
// MI455X (gfx1250) — matrix operand order repaired and checked by reading, not compiled, not run on hardware
//
#include <hip/hip_runtime.h>

#define Bb  2
#define Ss  2048
#define Dd  1024
#define Hh  16
#define HDd 64

typedef unsigned short ub16;
typedef unsigned int   u32;
typedef __attribute__((ext_vector_type(16))) __bf16 v16bf;
typedef __attribute__((ext_vector_type(8)))  float  v8f;
typedef __attribute__((ext_vector_type(4)))  unsigned int u32x4;
typedef __attribute__((ext_vector_type(8)))  int          i32x8;
typedef __attribute__((ext_vector_type(4)))  int          i32x4;

union Frag { v16bf v; u32 u[8]; };

__device__ __forceinline__ ub16 f2bf(float x) {
  u32 u = __float_as_uint(x);
  u += 0x7fffu + ((u >> 16) & 1u);   // round-to-nearest-even
  return (ub16)(u >> 16);
}

__device__ __forceinline__ v8f wmma_bf16(const Frag& a, const Frag& b, v8f c) {
  return __builtin_amdgcn_wmma_f32_16x16x32_bf16(
      /*neg_a=*/false, a.v, /*neg_b=*/false, b.v,
      /*c_mod=*/(short)0, c, /*reuse_a=*/false, /*reuse_b=*/false);
}

// TDM: async-load a 64x64 bf16 tile (row stride Dd elems) into LDS with a
// 72-ushort destination pitch (pad_interval=32 DWORDs, pad_amount=4 DWORDs).
// D# layout per CDNA5 ISA.
__device__ __forceinline__ void tdm_load_tile(const ub16* gsrc, ub16* ldst) {
  size_t ga = (size_t)gsrc;                 // global byte address
  u32    la = (u32)(size_t)ldst;            // LDS byte offset (addr[31:0])
  u32x4 g0;
  g0[0] = 1u;                               // count=1, user mode
  g0[1] = la;                               // lds_addr
  g0[2] = (u32)(ga & 0xffffffffu);          // global_addr[31:0]
  g0[3] = (u32)((ga >> 32) & 0x01ffffffu)   // global_addr[56:32]
        | (2u << 30);                       // type=2 ("image")
  i32x8 g1;
  g1[0] = (int)((1u << 16)                  // data_size=1 -> 2 bytes
        |       (1u << 20)                  // pad_enable
        |       (4u << 22)                  // pad_interval: 32 DWORDs (128B row)
        |       (3u << 25));                // pad_amount: 4 DWORDs (16B pad)
  g1[1] = (int)(64u << 16);                 // ab_addr=0 | tensor_dim0[15:0]=64
  g1[2] = (int)(64u << 16);                 // tensor_dim0[31:16]=0 | tensor_dim1[15:0]=64
  g1[3] = (int)(64u << 16);                 // tensor_dim1[31:16]=0 | tile_dim0=64
  g1[4] = (int)(64u);                       // tile_dim1=64 | tile_dim2=0
  g1[5] = (int)(Dd);                        // tensor_dim0_stride[31:0]=1024
  g1[6] = 0;                                // stride hi | tensor_dim1_stride lo
  g1[7] = 0;
  i32x4 g2 = {};                            // 2-D tensor: groups 2/3 unused
  i32x4 g3 = {};
#if __clang_major__ >= 23
  i32x8 g4 = {};
  __builtin_amdgcn_tensor_load_to_lds(g0, g1, g2, g3, g4, 0);
#else
  __builtin_amdgcn_tensor_load_to_lds(g0, g1, g2, g3, 0);
#endif
}

// ---------------------------------------------------------------- cvt f32->bf16
__global__ void cvt_bf16(const float* __restrict__ s, ub16* __restrict__ d, int n) {
  int i = blockIdx.x * 256 + threadIdx.x;
  if (i < n) d[i] = f2bf(s[i]);
}

// ---------------------------------------------------------------- GEMM: C = A (MxK) * Bw(NxK)^T + bias
// block: 256 threads = 8 waves (4 wave-rows x 2 wave-cols); tile 64(M) x 128(N), K-step 64
template<int OUT_BF16>
__global__ __launch_bounds__(256) void gemm_bias(
    const ub16* __restrict__ A, const ub16* __restrict__ Bw,
    const float* __restrict__ bias, float* __restrict__ Cf, ub16* __restrict__ Cb,
    int M, int N, int K) {
  __shared__ ub16 As[64 * 72];
  __shared__ ub16 Bs[128 * 72];
  const int tid  = threadIdx.x;
  const int lane = tid & 31, wave = tid >> 5;
  const int wr = wave >> 1, wc = wave & 1;
  const int hl = lane >> 4, ln = lane & 15;
  const int mblk = blockIdx.y * 64;
  const int nblk = blockIdx.x * 128;

  v8f acc[4];
  for (int g = 0; g < 4; ++g)
    for (int i = 0; i < 8; ++i) acc[g][i] = 0.f;

  for (int kk = 0; kk < K; kk += 64) {
    __syncthreads();
#pragma unroll
    for (int i = 0; i < 2; ++i) {          // A tile 64x64 bf16
      int idx = tid + i * 256;
      int row = idx >> 3, seg = idx & 7;
      *(uint4*)&As[row * 72 + seg * 8] =
          *(const uint4*)&A[(size_t)(mblk + row) * K + kk + seg * 8];
    }
#pragma unroll
    for (int i = 0; i < 4; ++i) {          // B tile 128x64 bf16 (N x K)
      int idx = tid + i * 256;
      int row = idx >> 3, seg = idx & 7;
      *(uint4*)&Bs[row * 72 + seg * 8] =
          *(const uint4*)&Bw[(size_t)(nblk + row) * K + kk + seg * 8];
    }
    __syncthreads();

    Frag af[2];
#pragma unroll
    for (int c = 0; c < 2; ++c) {
      const ub16* r = &As[(wr * 16 + ln) * 72 + c * 32];
#pragma unroll
      for (int v = 0; v < 8; ++v)
        af[c].u[v] = *(const u32*)&r[(hl << 4) + (v << 1)];
    }
#pragma unroll
    for (int g = 0; g < 4; ++g) {
#pragma unroll
      for (int c = 0; c < 2; ++c) {
        Frag bf;
        const ub16* r = &Bs[(wc * 64 + g * 16 + ln) * 72 + c * 32 + hl * 16];
#pragma unroll
        for (int v = 0; v < 8; ++v) bf.u[v] = *(const u32*)&r[v * 2];
        acc[g] = wmma_bf16(af[c], bf, acc[g]);
      }
    }
  }

#pragma unroll
  for (int g = 0; g < 4; ++g) {
    int col = nblk + wc * 64 + g * 16 + ln;
    float bv = bias[col];
#pragma unroll
    for (int i = 0; i < 8; ++i) {
      int row = mblk + wr * 16 + i + 8 * hl;
      float o = acc[g][i] + bv;
      if (OUT_BF16) Cb[(size_t)row * N + col] = f2bf(o);
      else          Cf[(size_t)row * N + col] = o;
    }
  }
}

// ---------------------------------------------------------------- flash attention
// grid: (S/64, B*H); block: 128 threads = 4 waves, each wave owns 16 q-rows.
// K/V tiles double-buffered in LDS, staged by the Tensor Data Mover (wave 0),
// overlapped with WMMA + online-softmax on the current tile.
__global__ __launch_bounds__(128) void flash_attn(
    const ub16* __restrict__ Qp, const ub16* __restrict__ Kp, const ub16* __restrict__ Vp,
    const float* __restrict__ pad, ub16* __restrict__ Ctx) {
  __shared__ ub16 Qs[64 * 72];
  __shared__ ub16 KVs[2][2][64 * 72];      // [buffer][K/V][tile]
  __shared__ ub16 Ps[4 * 16 * 72];
  const int tid  = threadIdx.x;
  const int lane = tid & 31, wave = tid >> 5;
  const int hl = lane >> 4, ln = lane & 15;
  const int bh = blockIdx.y, b = bh / Hh, h = bh % Hh;
  const int q0 = blockIdx.x * 64;
  const float scale = 0.125f;              // 1/sqrt(64)
  const size_t headoff = (size_t)h * HDd;
  const int nkt = blockIdx.x + 1;          // causal upper bound on k-tiles

  const ub16* Kbase = Kp + (size_t)b * Ss * Dd + headoff;
  const ub16* Vbase = Vp + (size_t)b * Ss * Dd + headoff;

  // prefetch k-tile 0 via TDM while we stage Q
  if (wave == 0) {
    tdm_load_tile(Kbase, &KVs[0][0][0]);
    tdm_load_tile(Vbase, &KVs[0][1][0]);
  }

#pragma unroll
  for (int i = 0; i < 4; ++i) {            // Q tile 64x64
    int idx = tid + i * 128;
    int row = idx >> 3, seg = idx & 7;
    *(uint4*)&Qs[row * 72 + seg * 8] =
        *(const uint4*)&Qp[((size_t)(b * Ss + q0 + row)) * Dd + headoff + seg * 8];
  }
  __syncthreads();

  Frag qf[2];
#pragma unroll
  for (int c = 0; c < 2; ++c) {
    const ub16* r = &Qs[(wave * 16 + ln) * 72 + c * 32];
#pragma unroll
    for (int v = 0; v < 8; ++v)
      qf[c].u[v] = *(const u32*)&r[(hl << 4) + (v << 1)];
  }

  float mrow[8], lrow[8];
  v8f cacc[4];
#pragma unroll
  for (int i = 0; i < 8; ++i) { mrow[i] = -3.0e38f; lrow[i] = 0.f; }
  for (int g = 0; g < 4; ++g)
    for (int i = 0; i < 8; ++i) cacc[g][i] = 0.f;

  ub16* Pw = &Ps[wave * 16 * 72];

  for (int kt = 0; kt < nkt; ++kt) {
    if (wave == 0) __builtin_amdgcn_s_wait_tensorcnt(0);  // tile kt landed
    __syncthreads();
    if (wave == 0 && kt + 1 < nkt) {       // kick off next tile into other buffer
      size_t off = (size_t)(kt + 1) * 64 * Dd;
      tdm_load_tile(Kbase + off, &KVs[(kt + 1) & 1][0][0]);
      tdm_load_tile(Vbase + off, &KVs[(kt + 1) & 1][1][0]);
    }
    const ub16* Ks = &KVs[kt & 1][0][0];
    const ub16* Vs = &KVs[kt & 1][1][0];

    // logits: 16 q x 64 keys per wave
    v8f sacc[4];
    for (int g = 0; g < 4; ++g)
      for (int i = 0; i < 8; ++i) sacc[g][i] = 0.f;
#pragma unroll
    for (int g = 0; g < 4; ++g)
#pragma unroll
      for (int c = 0; c < 2; ++c) {
        Frag kf;
        const ub16* r = Ks + (g * 16 + ln) * 72 + c * 32 + hl * 16;
#pragma unroll
        for (int v = 0; v < 8; ++v) kf.u[v] = *(const u32*)&r[v * 2];
        sacc[g] = wmma_bf16(qf[c], kf, sacc[g]);
      }

    // scale + padding mask + causal mask
#pragma unroll
    for (int g = 0; g < 4; ++g) {
      int col = kt * 64 + g * 16 + ln;
      float pv = pad[b * Ss + col];
#pragma unroll
      for (int i = 0; i < 8; ++i) {
        int qrow = q0 + wave * 16 + i + 8 * hl;
        float s = sacc[g][i] * scale + pv;
        sacc[g][i] = (col > qrow) ? -1.0e9f : s;
      }
    }

    // online softmax: row max across the 16-lane half
    float tm[8];
#pragma unroll
    for (int i = 0; i < 8; ++i) {
      float t = fmaxf(fmaxf(sacc[0][i], sacc[1][i]), fmaxf(sacc[2][i], sacc[3][i]));
#pragma unroll
      for (int off = 1; off < 16; off <<= 1)
        t = fmaxf(t, __shfl_xor(t, off, 32));
      tm[i] = t;
    }
#pragma unroll
    for (int i = 0; i < 8; ++i) {
      float mn = fmaxf(mrow[i], tm[i]);
      float alpha = __expf(mrow[i] - mn);
      mrow[i] = mn;
      lrow[i] *= alpha;
#pragma unroll
      for (int g = 0; g < 4; ++g) cacc[g][i] *= alpha;
    }

    // p = exp(s-m); row sums; P -> per-wave LDS (bf16) for the P@V WMMA
    float rs[8];
    for (int i = 0; i < 8; ++i) rs[i] = 0.f;
#pragma unroll
    for (int g = 0; g < 4; ++g)
#pragma unroll
      for (int i = 0; i < 8; ++i) {
        float p = __expf(sacc[g][i] - mrow[i]);
        rs[i] += p;
        Pw[(i + 8 * hl) * 72 + g * 16 + ln] = f2bf(p);
      }
#pragma unroll
    for (int i = 0; i < 8; ++i) {
      float t = rs[i];
#pragma unroll
      for (int off = 1; off < 16; off <<= 1) t += __shfl_xor(t, off, 32);
      lrow[i] += t;
    }

    asm volatile("s_wait_dscnt 0" ::: "memory");   // P stores visible to same-wave ds reads

    // ctx += P(16x64) @ V(64x64)
#pragma unroll
    for (int c = 0; c < 2; ++c) {
      Frag pf;
      const ub16* r = Pw + ln * 72 + c * 32;
#pragma unroll
      for (int v = 0; v < 8; ++v)
        pf.u[v] = *(const u32*)&r[(hl << 4) + (v << 1)];
#pragma unroll
      for (int g = 0; g < 4; ++g) {
        Frag vf;                                   // B operand read transposed from Vs
#pragma unroll
        for (int v = 0; v < 8; ++v) {
          int k = c * 32 + hl * 16 + v * 2;
          u32 lo = Vs[k * 72 + g * 16 + ln];
          u32 hi = Vs[(k + 1) * 72 + g * 16 + ln];
          vf.u[v] = lo | (hi << 16);
        }
        cacc[g] = wmma_bf16(pf, vf, cacc[g]);
      }
    }
  }

  // normalize + store ctx (bf16, [B*S, D] with head-interleaved cols)
#pragma unroll
  for (int g = 0; g < 4; ++g)
#pragma unroll
    for (int i = 0; i < 8; ++i) {
      int qrow = q0 + wave * 16 + i + 8 * hl;
      float o = cacc[g][i] / lrow[i];
      Ctx[((size_t)(b * Ss + qrow)) * Dd + headoff + g * 16 + ln] = f2bf(o);
    }
}

// ---------------------------------------------------------------- launch
extern "C" void kernel_launch(void* const* d_in, const int* in_sizes, int n_in,
                              void* d_out, int out_size, void* d_ws, size_t ws_size,
                              hipStream_t stream) {
  const float* q   = (const float*)d_in[0];
  const float* k   = (const float*)d_in[1];
  const float* v   = (const float*)d_in[2];
  const float* Wq  = (const float*)d_in[3];
  const float* bq  = (const float*)d_in[4];
  const float* Wk  = (const float*)d_in[5];
  const float* bk  = (const float*)d_in[6];
  const float* Wv  = (const float*)d_in[7];
  const float* bv  = (const float*)d_in[8];
  const float* Wo  = (const float*)d_in[9];
  const float* bo  = (const float*)d_in[10];
  // d_in[11] = attention_mask (causal, applied analytically in-kernel)
  const float* pad = (const float*)d_in[12];

  const int M = Bb * Ss;                  // 4096
  const int N = Dd, K = Dd;               // 1024
  const size_t nW = (size_t)Dd * Dd;      // 1M
  const size_t nX = (size_t)Bb * Ss * Dd; // 4M

  char* w = (char*)d_ws;
  ub16* Wq_b = (ub16*)w; w += nW * 2;
  ub16* Wk_b = (ub16*)w; w += nW * 2;
  ub16* Wv_b = (ub16*)w; w += nW * 2;
  ub16* Wo_b = (ub16*)w; w += nW * 2;
  ub16* xq_b = (ub16*)w; w += nX * 2;
  ub16* xk_b = (ub16*)w; w += nX * 2;
  ub16* xv_b = (ub16*)w; w += nX * 2;
  ub16* Qp   = (ub16*)w; w += nX * 2;
  ub16* Kp   = (ub16*)w; w += nX * 2;
  ub16* Vp   = (ub16*)w; w += nX * 2;
  ub16* Ctx  = (ub16*)w; w += nX * 2;

  int gW = (int)((nW + 255) / 256);
  int gX = (int)((nX + 255) / 256);
  cvt_bf16<<<gW, 256, 0, stream>>>(Wq, Wq_b, (int)nW);
  cvt_bf16<<<gW, 256, 0, stream>>>(Wk, Wk_b, (int)nW);
  cvt_bf16<<<gW, 256, 0, stream>>>(Wv, Wv_b, (int)nW);
  cvt_bf16<<<gW, 256, 0, stream>>>(Wo, Wo_b, (int)nW);
  cvt_bf16<<<gX, 256, 0, stream>>>(q, xq_b, (int)nX);
  cvt_bf16<<<gX, 256, 0, stream>>>(k, xk_b, (int)nX);
  cvt_bf16<<<gX, 256, 0, stream>>>(v, xv_b, (int)nX);

  dim3 ggrid(N / 128, M / 64);
  gemm_bias<1><<<ggrid, 256, 0, stream>>>(xq_b, Wq_b, bq, nullptr, Qp, M, N, K);
  gemm_bias<1><<<ggrid, 256, 0, stream>>>(xk_b, Wk_b, bk, nullptr, Kp, M, N, K);
  gemm_bias<1><<<ggrid, 256, 0, stream>>>(xv_b, Wv_b, bv, nullptr, Vp, M, N, K);

  flash_attn<<<dim3(Ss / 64, Bb * Hh), 128, 0, stream>>>(Qp, Kp, Vp, pad, Ctx);

  gemm_bias<0><<<ggrid, 256, 0, stream>>>(Ctx, Wo_b, bo, (float*)d_out, nullptr, M, N, K);
}
